// UTransformer_72593537237197
// MI455X (gfx1250) — matrix operand order repaired and checked by reading, not compiled, not run on hardware
//
#include <hip/hip_runtime.h>
#include <hip/hip_bf16.h>

// Flip to 0 if the gfx1250 assembler rejects the async-to-LDS mnemonics.
#define USE_ASYNC_STAGING 1

// ---------------------------------------------------------------------------
// CDNA5 WMMA types (wave32, v_wmma_f32_16x16x32_bf16)
// ---------------------------------------------------------------------------
typedef __attribute__((ext_vector_type(16))) __bf16 v16bf;
typedef __attribute__((ext_vector_type(8)))  float  v8f;

#define NE_   2048
#define DIM_  512
#define DFF_  2048
#define DK_   64
#define VOC_  32000
#define DEPTH_ 8

// ---------------------------------------------------------------------------
// Async global->LDS copy (GLOBAL_LOAD_ASYNC_TO_LDS_B128, ASYNCcnt-tracked).
// vdst = 32-bit LDS byte address (generic LDS pointer bits [31:0]),
// vaddr = 64-bit global address. Per-lane addressing, EXEC all-ones here.
// ---------------------------------------------------------------------------
__device__ __forceinline__ void async_load_b128(unsigned lds_addr, const void* g) {
    asm volatile("global_load_async_to_lds_b128 %0, %1, off"
                 :: "v"(lds_addr), "v"((unsigned long long)(size_t)g)
                 : "memory");
}
__device__ __forceinline__ void wait_async0() {
    asm volatile("s_wait_asynccnt 0x0" ::: "memory");
}
__device__ __forceinline__ unsigned lds_addr32(const void* p) {
    return (unsigned)(size_t)p;   // LDS aperture: byte offset lives in [31:0]
}

// ---------------------------------------------------------------------------
// Fragment loaders from LDS (contiguous k order, same for A and B).
// A 16x32: lane l (m=l&15, h=l>>4): elems 0..15 -> K=k0+h*16.. for row m.
// B 32x16: lane l (n=l&15, h=l>>4): elems 0..15 -> K=k0+h*16.. for column n.
// B tiles are stored transposed in LDS ([n][k]) so both loads are contiguous.
// ---------------------------------------------------------------------------
__device__ __forceinline__ v16bf load_a_frag(const __bf16* tile, int ldk,
                                             int mrow0, int k0, int lane) {
    int m = lane & 15, h = lane >> 4;
    const __bf16* p0 = tile + (size_t)(mrow0 + m) * ldk + k0 + h * 16;
    const __bf16* p1 = p0 + 8;
    v16bf a;
#pragma unroll
    for (int i = 0; i < 8; ++i) a[i] = p0[i];
#pragma unroll
    for (int i = 0; i < 8; ++i) a[8 + i] = p1[i];
    return a;
}

__device__ __forceinline__ v16bf load_b_frag(const __bf16* bt, int ldk,
                                             int ncol0, int k0, int lane) {
    int n = lane & 15, h = lane >> 4;
    const __bf16* p = bt + (size_t)(ncol0 + n) * ldk + k0 + h * 16;
    v16bf b;
#pragma unroll
    for (int i = 0; i < 16; ++i) b[i] = p[i];
    return b;
}

// ---------------------------------------------------------------------------
// Weight prep: f32 W[K,N] -> bf16 WT[N,K], hoisted out of all GEMMs.
// ---------------------------------------------------------------------------
__global__ __launch_bounds__(256) void transpose_bf16_kernel(
    const float* __restrict__ in, __bf16* __restrict__ out, int K, int N) {
    __shared__ float tile[32][33];
    const int bn = blockIdx.x * 32, bk = blockIdx.y * 32;
    const int tx = threadIdx.x & 31, ty = threadIdx.x >> 5;
#pragma unroll
    for (int r = ty; r < 32; r += 8)
        tile[r][tx] = in[(size_t)(bk + r) * N + bn + tx];
    __syncthreads();
#pragma unroll
    for (int r = ty; r < 32; r += 8)
        out[(size_t)(bn + r) * K + bk + tx] = (__bf16)tile[tx][r];
}

// ---------------------------------------------------------------------------
// GEMM: A bf16 [M,K] @ (WT bf16 [N,K])^T -> Cf f32 and/or Cb bf16 [M,N].
// Block tile 256x64, BK=32; 8 waves, wave owns 2 row-tiles x 4 col-tiles.
// Double-buffered LDS, async global->LDS staging: tile k+1 streams through
// the async engine while WMMAs consume tile k; one barrier per iteration.
// ---------------------------------------------------------------------------
__global__ __launch_bounds__(256) void gemm_bf16_kernel(
    const __bf16* __restrict__ A, const __bf16* __restrict__ WT,
    float* __restrict__ Cf, __bf16* __restrict__ Cb,
    int M, int N, int K, int relu) {
    __shared__ __align__(16) __bf16 As[2][256 * 32];   // 2 x 16KB
    __shared__ __align__(16) __bf16 Bs[2][64 * 32];    // 2 x 4KB

    const int n0 = blockIdx.x * 64;
    const int m0 = blockIdx.y * 256;
    const int tid = threadIdx.x, lane = tid & 31, wave = tid >> 5;

#if USE_ASYNC_STAGING
    const unsigned asb = lds_addr32(&As[0][0]);
    const unsigned bsb = lds_addr32(&Bs[0][0]);
    auto issue_tile = [&](int kt) {
        const unsigned buf = (unsigned)(kt & 1);
        const __bf16* abase = A + (size_t)m0 * K + kt * 32;
#pragma unroll
        for (int i = 0; i < 4; ++i) {           // A: 1024 16B chunks, 4/thread
            int e = tid + i * 256;              // e*16B == (r*32 + cc*8) elems
            int r = e >> 2, cc = e & 3;
            async_load_b128(asb + buf * 16384u + (unsigned)e * 16u,
                            abase + (size_t)r * K + cc * 8);
        }
        int r = tid >> 2, cc = tid & 3;         // B: 256 16B chunks, 1/thread
        async_load_b128(bsb + buf * 4096u + (unsigned)tid * 16u,
                        WT + (size_t)(n0 + r) * K + kt * 32 + cc * 8);
    };
#else
    int4 ra[4], rb;
    auto issue_tile = [&](int kt) {
        const __bf16* abase = A + (size_t)m0 * K + kt * 32;
#pragma unroll
        for (int i = 0; i < 4; ++i) {
            int e = tid + i * 256;
            int r = e >> 2, cc = e & 3;
            ra[i] = *(const int4*)(abase + (size_t)r * K + cc * 8);
        }
        int r = tid >> 2, cc = tid & 3;
        rb = *(const int4*)(WT + (size_t)(n0 + r) * K + kt * 32 + cc * 8);
    };
#endif

    v8f acc[2][4];
#pragma unroll
    for (int rt = 0; rt < 2; ++rt)
#pragma unroll
        for (int j = 0; j < 4; ++j) acc[rt][j] = v8f{};

    const int nt = K >> 5;
    issue_tile(0);
    for (int kt = 0; kt < nt; ++kt) {
#if USE_ASYNC_STAGING
        wait_async0();                   // this wave's tile-kt data is in LDS
        __syncthreads();                 // all waves' data in; prior reads done
        if (kt + 1 < nt) issue_tile(kt + 1);   // streams during WMMAs below
        const __bf16* as = As[kt & 1];
        const __bf16* bs = Bs[kt & 1];
#else
        __syncthreads();
        {
#pragma unroll
            for (int i = 0; i < 4; ++i) {
                int e = tid + i * 256;
                *(int4*)(&As[0][0] + (size_t)e * 8) = ra[i];
            }
            *(int4*)(&Bs[0][0] + (size_t)tid * 8) = rb;
        }
        __syncthreads();
        if (kt + 1 < nt) issue_tile(kt + 1);
        const __bf16* as = As[0];
        const __bf16* bs = Bs[0];
#endif
        v16bf bf[4];
#pragma unroll
        for (int j = 0; j < 4; ++j) bf[j] = load_b_frag(bs, 32, j * 16, 0, lane);
#pragma unroll
        for (int rt = 0; rt < 2; ++rt) {
            v16bf a = load_a_frag(as, 32, wave * 32 + rt * 16, 0, lane);
#pragma unroll
            for (int j = 0; j < 4; ++j)
                acc[rt][j] = __builtin_amdgcn_wmma_f32_16x16x32_bf16(
                    false, a, false, bf[j], (short)0, acc[rt][j], false, false);
        }
    }

    const int n = lane & 15, hr = lane >> 4;
#pragma unroll
    for (int rt = 0; rt < 2; ++rt)
#pragma unroll
        for (int j = 0; j < 4; ++j)
#pragma unroll
            for (int r = 0; r < 8; ++r) {
                int row = m0 + wave * 32 + rt * 16 + hr * 8 + r;
                int col = n0 + j * 16 + n;
                float v = acc[rt][j][r];
                if (relu) v = fmaxf(v, 0.0f);
                if (Cf) Cf[(size_t)row * N + col] = v;
                if (Cb) Cb[(size_t)row * N + col] = (__bf16)v;
            }
}

// ---------------------------------------------------------------------------
// Dense per-(sentence, head) attention over bf16 inputs:
//   S = exp((Q K^T)/8) (causal optional), O = (S V) / rowsum(S)  -> bf16
// One block per (b,h). 256 threads = 8 waves; wave w owns S row-tile w.
// Q/K staged via async global->LDS; V via VGPRs (needs per-element transpose).
// ---------------------------------------------------------------------------
__global__ __launch_bounds__(256) void attn_kernel(
    const __bf16* __restrict__ Qb, int ldq,
    const __bf16* __restrict__ Kb, int ldk,
    const __bf16* __restrict__ Vb, int ldv,
    __bf16* __restrict__ Ob, int ldo, int causal) {
    __shared__ __align__(16) __bf16 smem[3 * 8192];
    __shared__ float rowsum[128];
    __bf16* Qs = smem;              // [128][64]
    __bf16* Ks = smem + 8192;       // [128][64] (B^T layout for S = Q K^T)
    __bf16* Vt = smem + 16384;      // [64][128] (B^T layout for O = S V)
    __bf16* Ss = smem;              // [128][128], reuses Qs+Ks after pass 1

    const int b = blockIdx.x >> 3, h = blockIdx.x & 7;
    const int tid = threadIdx.x, lane = tid & 31, wave = tid >> 5;

    const __bf16* Q = Qb + (size_t)(b * 128) * ldq + h * DK_;
    const __bf16* Kp = Kb + (size_t)(b * 128) * ldk + h * DK_;
    const __bf16* V = Vb + (size_t)(b * 128) * ldv + h * DK_;

#if USE_ASYNC_STAGING
    const unsigned qsb = lds_addr32(Qs), ksb = lds_addr32(Ks);
#endif
#pragma unroll
    for (int i = 0; i < 4; ++i) {           // 1024 16B chunks per matrix
        int e = tid + i * 256;
        int r = e >> 3, cc = e & 7;         // e*16B == (r*64 + cc*8) elems
#if USE_ASYNC_STAGING
        async_load_b128(qsb + (unsigned)e * 16u, Q + (size_t)r * ldq + cc * 8);
        async_load_b128(ksb + (unsigned)e * 16u, Kp + (size_t)r * ldk + cc * 8);
#else
        *(int4*)(Qs + (size_t)e * 8) = *(const int4*)(Q + (size_t)r * ldq + cc * 8);
        *(int4*)(Ks + (size_t)e * 8) = *(const int4*)(Kp + (size_t)r * ldk + cc * 8);
#endif
        int4 vv = *(const int4*)(V + (size_t)r * ldv + cc * 8);
        const __bf16* vp = (const __bf16*)&vv;
#pragma unroll
        for (int j = 0; j < 8; ++j) Vt[(cc * 8 + j) * 128 + r] = vp[j];
    }
#if USE_ASYNC_STAGING
    wait_async0();
#endif
    __syncthreads();

    // ---- pass 1: S = Q K^T (128x128, K-dim 64) ----
    v8f sacc[8];
#pragma unroll
    for (int j = 0; j < 8; ++j) sacc[j] = v8f{};
#pragma unroll
    for (int k0 = 0; k0 < 64; k0 += 32) {
        v16bf a = load_a_frag(Qs, 64, wave * 16, k0, lane);
#pragma unroll
        for (int j = 0; j < 8; ++j) {
            v16bf bf = load_b_frag(Ks, 64, j * 16, k0, lane);
            sacc[j] = __builtin_amdgcn_wmma_f32_16x16x32_bf16(
                false, a, false, bf, (short)0, sacc[j], false, false);
        }
    }
    __syncthreads();    // all waves done with Qs/Ks before Ss overwrite

    const int n = lane & 15, hr = lane >> 4;
#pragma unroll
    for (int j = 0; j < 8; ++j)
#pragma unroll
        for (int r = 0; r < 8; ++r) {
            int row = wave * 16 + hr * 8 + r;
            int col = j * 16 + n;
            float e = (causal && col > row) ? 0.0f : __expf(0.125f * sacc[j][r]);
            Ss[row * 128 + col] = (__bf16)e;
        }
    __syncthreads();

    if (tid < 128) {                        // rowsum; EXEC restored after
        float s = 0.0f;
        for (int c = 0; c < 128; ++c) s += (float)Ss[tid * 128 + c];
        rowsum[tid] = s;
    }
    __syncthreads();

    // ---- pass 2: O = S V (128x64, K-dim 128) ----
    v8f oacc[4];
#pragma unroll
    for (int j = 0; j < 4; ++j) oacc[j] = v8f{};
#pragma unroll
    for (int k0 = 0; k0 < 128; k0 += 32) {
        v16bf a = load_a_frag(Ss, 128, wave * 16, k0, lane);
#pragma unroll
        for (int j = 0; j < 4; ++j) {
            v16bf bf = load_b_frag(Vt, 128, j * 16, k0, lane);
            oacc[j] = __builtin_amdgcn_wmma_f32_16x16x32_bf16(
                false, a, false, bf, (short)0, oacc[j], false, false);
        }
    }

    __bf16* O = Ob + (size_t)(b * 128) * ldo + h * DK_;
#pragma unroll
    for (int j = 0; j < 4; ++j)
#pragma unroll
        for (int r = 0; r < 8; ++r) {
            int row = wave * 16 + hr * 8 + r;
            int col = j * 16 + n;
            O[(size_t)row * ldo + col] =
                (__bf16)(oacc[j][r] / (rowsum[row] + 1e-9f));
        }
}

// ---------------------------------------------------------------------------
__global__ __launch_bounds__(256) void embed_kernel(
    const int* __restrict__ tok, const int* __restrict__ pos,
    const float* __restrict__ emb, const float* __restrict__ pose,
    float* __restrict__ out) {
    int e = blockIdx.x * 256 + threadIdx.x;
    int nrow = e >> 9, d = e & 511;
    out[e] = emb[(size_t)tok[nrow] * DIM_ + d] + pose[(size_t)pos[nrow] * DIM_ + d];
}

__global__ __launch_bounds__(256) void add3_kernel(
    const float* __restrict__ x, const int* __restrict__ pos,
    const float* __restrict__ pose, const float* __restrict__ trow,
    __bf16* __restrict__ out) {
    int e = blockIdx.x * 256 + threadIdx.x;
    int nrow = e >> 9, d = e & 511;
    out[e] = (__bf16)(x[e] + pose[(size_t)pos[nrow] * DIM_ + d] + trow[d]);
}

__global__ __launch_bounds__(256) void ln_residual_kernel(
    float* __restrict__ x, const float* __restrict__ g,
    __bf16* __restrict__ xb) {
    __shared__ float red[256];
    const int tid = threadIdx.x;
    const size_t base = (size_t)blockIdx.x * DIM_;
    float a0 = x[base + tid] + g[base + tid];
    float a1 = x[base + tid + 256] + g[base + tid + 256];

    red[tid] = a0 + a1;
    __syncthreads();
    for (int off = 128; off > 0; off >>= 1) {
        if (tid < off) red[tid] += red[tid + off];
        __syncthreads();
    }
    const float mean = red[0] * (1.0f / DIM_);
    __syncthreads();

    const float d0 = a0 - mean, d1 = a1 - mean;
    red[tid] = d0 * d0 + d1 * d1;
    __syncthreads();
    for (int off = 128; off > 0; off >>= 1) {
        if (tid < off) red[tid] += red[tid + off];
        __syncthreads();
    }
    const float rstd = rsqrtf(red[0] * (1.0f / DIM_) + 1e-5f);
    const float o0 = d0 * rstd, o1 = d1 * rstd;
    x[base + tid] = o0;
    x[base + tid + 256] = o1;
    xb[base + tid] = (__bf16)o0;
    xb[base + tid + 256] = (__bf16)o1;
}

__global__ __launch_bounds__(256) void log_softmax_kernel(float* __restrict__ x) {
    __shared__ float red[256];
    const int tid = threadIdx.x;
    float* p = x + (size_t)blockIdx.x * VOC_;

    float mx = -3.4e38f;
    for (int i = tid; i < VOC_; i += 256) mx = fmaxf(mx, p[i]);
    red[tid] = mx;
    __syncthreads();
    for (int off = 128; off > 0; off >>= 1) {
        if (tid < off) red[tid] = fmaxf(red[tid], red[tid + off]);
        __syncthreads();
    }
    const float m = red[0];
    __syncthreads();

    float s = 0.0f;
    for (int i = tid; i < VOC_; i += 256) s += __expf(p[i] - m);
    red[tid] = s;
    __syncthreads();
    for (int off = 128; off > 0; off >>= 1) {
        if (tid < off) red[tid] += red[tid + off];
        __syncthreads();
    }
    const float lse = m + __logf(red[0]);
    __syncthreads();

    for (int i = tid; i < VOC_; i += 256) p[i] -= lse;
}

// ---------------------------------------------------------------------------
// Host orchestration
// ---------------------------------------------------------------------------
static inline void gemm(const __bf16* A, const __bf16* WT, float* Cf, __bf16* Cb,
                        int M, int N, int K, int relu, hipStream_t s) {
    gemm_bf16_kernel<<<dim3(N / 64, M / 256), 256, 0, s>>>(A, WT, Cf, Cb, M, N, K, relu);
}
static inline void wtrans(const float* W, __bf16* WT, int K, int N, hipStream_t s) {
    transpose_bf16_kernel<<<dim3(N / 32, K / 32), 256, 0, s>>>(W, WT, K, N);
}

extern "C" void kernel_launch(void* const* d_in, const int* in_sizes, int n_in,
                              void* d_out, int out_size, void* d_ws, size_t ws_size,
                              hipStream_t stream) {
    const float* emb_src  = (const float*)d_in[0];
    const float* emb_tgt  = (const float*)d_in[1];
    const float* pos_emb  = (const float*)d_in[2];
    const float* time_emb = (const float*)d_in[3];
    const float* Wqkv_e   = (const float*)d_in[4];
    const float* Wo_e     = (const float*)d_in[5];
    const float* W1_e     = (const float*)d_in[6];
    const float* W2_e     = (const float*)d_in[7];
    const float* Wqkv_d   = (const float*)d_in[8];
    const float* Wo_d     = (const float*)d_in[9];
    const float* W1_d     = (const float*)d_in[10];
    const float* W2_d     = (const float*)d_in[11];
    const float* Wq_x     = (const float*)d_in[12];
    const float* Wkv_x    = (const float*)d_in[13];
    const float* Wo_x     = (const float*)d_in[14];
    const float* Wgen     = (const float*)d_in[15];
    const int* src_tokens = (const int*)d_in[16];
    const int* tgt_tokens = (const int*)d_in[17];
    const int* src_pos    = (const int*)d_in[18];
    const int* tgt_pos    = (const int*)d_in[19];
    (void)in_sizes; (void)n_in; (void)out_size; (void)ws_size;

    char* wsp = (char*)d_ws;
    auto alloc_f = [&](size_t nn) { float* p = (float*)wsp; wsp += nn * 4; return p; };
    auto alloc_b = [&](size_t nn) { __bf16* p = (__bf16*)wsp; wsp += nn * 2; return p; };
    const size_t SZX = (size_t)NE_ * DIM_;

    float*  x_e   = alloc_f(SZX);
    float*  x_d   = alloc_f(SZX);
    float*  gbuf  = alloc_f(SZX);
    __bf16* xbf_e = alloc_b(SZX);
    __bf16* xbf_d = alloc_b(SZX);
    __bf16* xp    = alloc_b(SZX);
    __bf16* attnb = alloc_b(SZX);
    __bf16* qkv   = alloc_b(3 * SZX);
    __bf16* hbuf  = alloc_b(4 * SZX);
    __bf16* kvx   = alloc_b(2 * SZX);
    __bf16* qx    = alloc_b(SZX);
    __bf16* WTqkv_e = alloc_b((size_t)DIM_ * 3 * DIM_);
    __bf16* WTo_e   = alloc_b((size_t)DIM_ * DIM_);
    __bf16* WT1_e   = alloc_b((size_t)DIM_ * DFF_);
    __bf16* WT2_e   = alloc_b((size_t)DFF_ * DIM_);
    __bf16* WTqkv_d = alloc_b((size_t)DIM_ * 3 * DIM_);
    __bf16* WTo_d   = alloc_b((size_t)DIM_ * DIM_);
    __bf16* WT1_d   = alloc_b((size_t)DIM_ * DFF_);
    __bf16* WT2_d   = alloc_b((size_t)DFF_ * DIM_);
    __bf16* WTq_x   = alloc_b((size_t)DIM_ * DIM_);
    __bf16* WTkv_x  = alloc_b((size_t)DIM_ * 2 * DIM_);
    __bf16* WTo_x   = alloc_b((size_t)DIM_ * DIM_);
    __bf16* WTgen   = alloc_b((size_t)DIM_ * VOC_);

    float* logits = (float*)d_out;            // [2048, 32000] fp32

    wtrans(Wqkv_e, WTqkv_e, DIM_, 3 * DIM_, stream);
    wtrans(Wo_e,   WTo_e,   DIM_, DIM_,     stream);
    wtrans(W1_e,   WT1_e,   DIM_, DFF_,     stream);
    wtrans(W2_e,   WT2_e,   DFF_, DIM_,     stream);
    wtrans(Wqkv_d, WTqkv_d, DIM_, 3 * DIM_, stream);
    wtrans(Wo_d,   WTo_d,   DIM_, DIM_,     stream);
    wtrans(W1_d,   WT1_d,   DIM_, DFF_,     stream);
    wtrans(W2_d,   WT2_d,   DFF_, DIM_,     stream);
    wtrans(Wq_x,   WTq_x,   DIM_, DIM_,     stream);
    wtrans(Wkv_x,  WTkv_x,  DIM_, 2 * DIM_, stream);
    wtrans(Wo_x,   WTo_x,   DIM_, DIM_,     stream);
    wtrans(Wgen,   WTgen,   DIM_, VOC_,     stream);

    const int EW = (NE_ * DIM_) / 256;

    embed_kernel<<<EW, 256, 0, stream>>>(src_tokens, src_pos, emb_src, pos_emb, x_e);
    embed_kernel<<<EW, 256, 0, stream>>>(tgt_tokens, tgt_pos, emb_tgt, pos_emb, x_d);

    // ---- encoder ----
    for (int t = 0; t < DEPTH_; ++t) {
        const float* trow = time_emb + (size_t)t * DIM_;
        add3_kernel<<<EW, 256, 0, stream>>>(x_e, src_pos, pos_emb, trow, xp);
        gemm(xp, WTqkv_e, nullptr, qkv, NE_, 3 * DIM_, DIM_, 0, stream);
        attn_kernel<<<128, 256, 0, stream>>>(qkv, 3 * DIM_, qkv + DIM_, 3 * DIM_,
                                             qkv + 2 * DIM_, 3 * DIM_,
                                             attnb, DIM_, 0);
        gemm(attnb, WTo_e, gbuf, nullptr, NE_, DIM_, DIM_, 0, stream);
        ln_residual_kernel<<<NE_, 256, 0, stream>>>(x_e, gbuf, xbf_e);
        gemm(xbf_e, WT1_e, nullptr, hbuf, NE_, DFF_, DIM_, 1, stream);
        gemm(hbuf, WT2_e, gbuf, nullptr, NE_, DIM_, DFF_, 0, stream);
        ln_residual_kernel<<<NE_, 256, 0, stream>>>(x_e, gbuf, xbf_e);
    }

    gemm(xbf_e, WTkv_x, nullptr, kvx, NE_, 2 * DIM_, DIM_, 0, stream);

    // ---- decoder ----
    for (int t = 0; t < DEPTH_; ++t) {
        const float* trow = time_emb + (size_t)t * DIM_;
        add3_kernel<<<EW, 256, 0, stream>>>(x_d, tgt_pos, pos_emb, trow, xp);
        gemm(xp, WTqkv_d, nullptr, qkv, NE_, 3 * DIM_, DIM_, 0, stream);
        attn_kernel<<<128, 256, 0, stream>>>(qkv, 3 * DIM_, qkv + DIM_, 3 * DIM_,
                                             qkv + 2 * DIM_, 3 * DIM_,
                                             attnb, DIM_, 1);
        gemm(attnb, WTo_d, gbuf, nullptr, NE_, DIM_, DIM_, 0, stream);
        ln_residual_kernel<<<NE_, 256, 0, stream>>>(x_d, gbuf, xbf_d);

        add3_kernel<<<EW, 256, 0, stream>>>(x_d, tgt_pos, pos_emb, trow, xp);
        gemm(xp, WTq_x, nullptr, qx, NE_, DIM_, DIM_, 0, stream);
        attn_kernel<<<128, 256, 0, stream>>>(qx, DIM_, kvx, 2 * DIM_,
                                             kvx + DIM_, 2 * DIM_,
                                             attnb, DIM_, 0);
        gemm(attnb, WTo_x, gbuf, nullptr, NE_, DIM_, DIM_, 0, stream);
        ln_residual_kernel<<<NE_, 256, 0, stream>>>(x_d, gbuf, xbf_d);

        gemm(xbf_d, WT1_d, nullptr, hbuf, NE_, DFF_, DIM_, 1, stream);
        gemm(hbuf, WT2_d, gbuf, nullptr, NE_, DIM_, DFF_, 0, stream);
        ln_residual_kernel<<<NE_, 256, 0, stream>>>(x_d, gbuf, xbf_d);
    }

    gemm(xbf_d, WTgen, logits, nullptr, NE_, VOC_, DIM_, 0, stream);
    log_softmax_kernel<<<NE_, 256, 0, stream>>>(logits);
}
